// CausalAttention_58179626991886
// MI455X (gfx1250) — matrix operand order repaired and checked by reading, not compiled, not run on hardware
//
#include <hip/hip_runtime.h>

// ---------- vector types ----------
typedef __attribute__((ext_vector_type(16))) _Float16 v16h;
typedef __attribute__((ext_vector_type(8)))  float    v8f;
typedef __attribute__((ext_vector_type(4)))  float    v4f;
typedef __attribute__((ext_vector_type(4)))  _Float16 v4h;
typedef __attribute__((ext_vector_type(4)))  unsigned int v4u;
typedef __attribute__((ext_vector_type(8)))  unsigned int v8u;

union FragU { v16h h; v4u u[2]; };

// ---------- WMMA fragment loaders (ISA 7.12.2 layouts, wave32) ----------
// A (16x32 f16): lane = {m(0..15), khalf}; VGPR0-3 = K[kh..kh+7], VGPR4-7 = K[16+kh..16+kh+7]
__device__ __forceinline__ v16h ld_fragA(const _Float16* p0, int ld, int row0) {
  const int lane = threadIdx.x & 31;
  const int r  = lane & 15;
  const int kh = (lane >> 4) << 3;            // 0 or 8
  const _Float16* p = p0 + (row0 + r) * ld + kh;
  FragU f;
  f.u[0] = *(const v4u*)(p);                  // K = kh .. kh+7
  f.u[1] = *(const v4u*)(p + 16);             // K = 16+kh .. 16+kh+7
  return f.h;
}
// B (32x16 f16) loaded from B^T stored N-major in LDS:
// lane = {n(0..15), khalf}; VGPR0-3 = K[kh..kh+7], VGPR4-7 = K[16+kh..16+kh+7] (same as A)
__device__ __forceinline__ v16h ld_fragB(const _Float16* p0, int ld, int row0) {
  const int lane = threadIdx.x & 31;
  const int n    = lane & 15;
  const int koff = (lane >> 4) << 3;          // 0 or 8
  const _Float16* p = p0 + (row0 + n) * ld + koff;
  FragU f;
  f.u[0] = *(const v4u*)(p);                  // K = koff .. koff+7
  f.u[1] = *(const v4u*)(p + 16);             // K = 16+koff .. 16+koff+7
  return f.h;
}

// 64x64 per wave: 4x4 tiles of 16x16, one K=32 step
__device__ __forceinline__ void mma_step(const _Float16* lA, const _Float16* lB,
                                         v8f acc[4][4], int wm, int wn) {
  v16h a[4], b[4];
#pragma unroll
  for (int i = 0; i < 4; ++i) a[i] = ld_fragA(lA, 32, wm + 16 * i);
#pragma unroll
  for (int i = 0; i < 4; ++i) b[i] = ld_fragB(lB, 32, wn + 16 * i);
#pragma unroll
  for (int mi = 0; mi < 4; ++mi)
#pragma unroll
    for (int ni = 0; ni < 4; ++ni)
      acc[mi][ni] = __builtin_amdgcn_wmma_f32_16x16x32_f16(
          false, a[mi], false, b[ni], (short)0, acc[mi][ni], false, false);
}

// ---------- Tensor Data Mover: 2D tile (tile_w x tile_h f16, row-major) -> LDS ----------
// D# per CDNA5 ISA; 2-group form (tensors up to 2D).
__device__ __forceinline__ unsigned lds_off(const void* p) {
  // generic LDS pointer = {aperture_hi, lds_offset}; low 32 bits are the LDS byte offset
  return (unsigned)(unsigned long long)p;
}
__device__ __forceinline__ void tdm_load_tile_f16(unsigned ldsAddr, const _Float16* gtile,
                                                  unsigned tensor_w, unsigned tensor_h,
                                                  unsigned stride_elems,
                                                  unsigned tile_w, unsigned tile_h) {
  const unsigned long long ga = (unsigned long long)gtile;
  v4u g0;
  g0.x = 1u;                                              // count=1, user mode, no gather
  g0.y = ldsAddr;                                         // lds_addr (bytes)
  g0.z = (unsigned)ga;                                    // global_addr[31:0]
  g0.w = (unsigned)((ga >> 32) & 0x01FFFFFFu) | (2u << 30); // global_addr[56:32] | type=2
  v8u g1;
  g1.s0 = 1u << 16;                                       // wg_mask=0, data_size=1 (2B), no flags
  g1.s1 = (tensor_w & 0xFFFFu) << 16;                     // tensor_dim0[15:0] @ bits 63:48
  g1.s2 = (tensor_w >> 16) | ((tensor_h & 0xFFFFu) << 16);// tensor_dim0[31:16] | tensor_dim1[15:0]
  g1.s3 = (tensor_h >> 16) | (tile_w << 16);              // tensor_dim1[31:16] | tile_dim0
  g1.s4 = tile_h;                                         // tile_dim1 (tile_dim2 = 0)
  g1.s5 = stride_elems;                                   // tensor_dim0_stride[31:0]
  g1.s6 = 0u;                                             // stride hi / dim1_stride lo
  g1.s7 = 0u;                                             // dim1_stride hi
  asm volatile("tensor_load_to_lds %0, %1" :: "s"(g0), "s"(g1) : "memory");
}

// ---------- kernel 1: X f32 -> f16 ----------
__global__ __launch_bounds__(256) void kconvert_x(const float* __restrict__ X,
                                                  _Float16* __restrict__ Xh, int n4) {
  int i = blockIdx.x * 256 + threadIdx.x;
  if (i >= n4) return;
  v4f x = ((const v4f*)X)[i];
  v4h h;
#pragma unroll
  for (int j = 0; j < 4; ++j) h[j] = (_Float16)x[j];
  ((v4h*)Xh)[i] = h;
}

// ---------- kernel 2: W [K][N] f32 -> W^T [N][K] f16 (tiled LDS transpose) ----------
__global__ __launch_bounds__(256) void kconvert_wT(const float* __restrict__ W0,
                                                   const float* __restrict__ W1,
                                                   const float* __restrict__ W2,
                                                   _Float16* __restrict__ Wt) {
  const float* Ws[3] = {W0, W1, W2};
  const float* W = Ws[blockIdx.z];
  _Float16* T = Wt + (size_t)blockIdx.z * 1024 * 1024;
  __shared__ _Float16 tile[32][33];
  const int k0 = blockIdx.y * 32, n0 = blockIdx.x * 32;
  const int c = threadIdx.x & 31, r0 = threadIdx.x >> 5;   // 8 rows / pass
#pragma unroll
  for (int i = 0; i < 4; ++i) {
    int r = r0 + 8 * i;
    tile[r][c] = (_Float16)W[(size_t)(k0 + r) * 1024 + n0 + c];
  }
  __syncthreads();
#pragma unroll
  for (int i = 0; i < 4; ++i) {
    int r = r0 + 8 * i;
    T[(size_t)(n0 + r) * 1024 + k0 + c] = tile[c][r];
  }
}

// ---------- kernel 3: QKV projection, TDM double-buffered ----------
// z=0 -> Q row-major, z=1 -> K row-major, z=2 -> V transposed per batch [b][d][kv]
__global__ __launch_bounds__(128) void kgemm_qkv(const _Float16* __restrict__ Xh,
                                                 const _Float16* __restrict__ Wt,
                                                 _Float16* __restrict__ QK,
                                                 _Float16* __restrict__ Vt) {
  __shared__ __align__(16) _Float16 lA[2][128 * 32];
  __shared__ __align__(16) _Float16 lB[2][128 * 32];
  const int m0 = blockIdx.y * 128;
  const int n0 = blockIdx.x * 128;
  const _Float16* Bt = Wt + (size_t)blockIdx.z * 1024 * 1024;
  const int t = threadIdx.x;
  const int w = t >> 5, wm = (w & 1) * 64, wn = (w >> 1) * 64;
  const bool issuer = (t < 32);                 // wave 0 drives the TDM (EXEC ignored by TDM)

  if (issuer) {                                 // prologue: tile k=0 -> buffer 0
    tdm_load_tile_f16(lds_off(&lA[0][0]), &Xh[(size_t)m0 * 1024], 1024, 8192, 1024, 32, 128);
    tdm_load_tile_f16(lds_off(&lB[0][0]), &Bt[(size_t)n0 * 1024], 1024, 1024, 1024, 32, 128);
    __builtin_amdgcn_s_wait_tensorcnt(0);
  }
  __syncthreads();

  v8f acc[4][4] = {};
  for (int k0 = 0; k0 < 1024; k0 += 32) {
    const int cur = (k0 >> 5) & 1, nxt = cur ^ 1;
    if (issuer && (k0 + 32 < 1024)) {           // DMA next tile while we do the math
      tdm_load_tile_f16(lds_off(&lA[nxt][0]), &Xh[(size_t)m0 * 1024 + k0 + 32], 1024, 8192, 1024, 32, 128);
      tdm_load_tile_f16(lds_off(&lB[nxt][0]), &Bt[(size_t)n0 * 1024 + k0 + 32], 1024, 1024, 1024, 32, 128);
    }
    mma_step(lA[cur], lB[cur], acc, wm, wn);
    if (issuer && (k0 + 32 < 1024)) __builtin_amdgcn_s_wait_tensorcnt(0);
    __syncthreads();                            // publish next buffer / retire reads of cur
  }

  const int lane = t & 31, nl = lane & 15, moff = (lane >> 4) * 8;
  if (blockIdx.z < 2) {
    _Float16* Out = QK + (size_t)blockIdx.z * 8192 * 1024;
#pragma unroll
    for (int mi = 0; mi < 4; ++mi)
#pragma unroll
      for (int ni = 0; ni < 4; ++ni) {
        int col = n0 + wn + 16 * ni + nl;
#pragma unroll
        for (int r = 0; r < 8; ++r) {
          int row = m0 + wm + 16 * mi + moff + r;
          Out[(size_t)row * 1024 + col] = (_Float16)acc[mi][ni][r];
        }
      }
  } else {                                      // V: store transposed [b][d][kv]
#pragma unroll
    for (int mi = 0; mi < 4; ++mi)
#pragma unroll
      for (int ni = 0; ni < 4; ++ni) {
        int col = n0 + wn + 16 * ni + nl;       // d
#pragma unroll
        for (int r = 0; r < 8; ++r) {
          int row = m0 + wm + 16 * mi + moff + r;   // b*2048 + kv
          Vt[(((size_t)(row >> 11)) << 21) + (size_t)col * 2048 + (row & 2047)] =
              (_Float16)acc[mi][ni][r];
        }
      }
  }
}

// ---------- kernel 4: scores = (Q K^T)/sqrt(d) as f16, causal blocks, TDM double-buffered ----------
__global__ __launch_bounds__(128) void kgemm_scores(const _Float16* __restrict__ Qh,
                                                    const _Float16* __restrict__ Kh,
                                                    _Float16* __restrict__ Sc) {
  if (blockIdx.x > blockIdx.y) return;   // fully above diagonal; never read by softmax
  __shared__ __align__(16) _Float16 lA[2][128 * 32];
  __shared__ __align__(16) _Float16 lB[2][128 * 32];
  const size_t boff = (size_t)blockIdx.z * 2048 * 1024;
  const _Float16* Qb = Qh + boff;
  const _Float16* Kb = Kh + boff;
  _Float16* Sb = Sc + (size_t)blockIdx.z * 2048 * 2048;
  const int n0 = blockIdx.x * 128;   // kv
  const int m0 = blockIdx.y * 128;   // q
  const int t = threadIdx.x;
  const int w = t >> 5, wm = (w & 1) * 64, wn = (w >> 1) * 64;
  const bool issuer = (t < 32);

  if (issuer) {
    tdm_load_tile_f16(lds_off(&lA[0][0]), &Qb[(size_t)m0 * 1024], 1024, 2048, 1024, 32, 128);
    tdm_load_tile_f16(lds_off(&lB[0][0]), &Kb[(size_t)n0 * 1024], 1024, 2048, 1024, 32, 128);
    __builtin_amdgcn_s_wait_tensorcnt(0);
  }
  __syncthreads();

  v8f acc[4][4] = {};
  for (int k0 = 0; k0 < 1024; k0 += 32) {
    const int cur = (k0 >> 5) & 1, nxt = cur ^ 1;
    if (issuer && (k0 + 32 < 1024)) {
      tdm_load_tile_f16(lds_off(&lA[nxt][0]), &Qb[(size_t)m0 * 1024 + k0 + 32], 1024, 2048, 1024, 32, 128);
      tdm_load_tile_f16(lds_off(&lB[nxt][0]), &Kb[(size_t)n0 * 1024 + k0 + 32], 1024, 2048, 1024, 32, 128);
    }
    mma_step(lA[cur], lB[cur], acc, wm, wn);
    if (issuer && (k0 + 32 < 1024)) __builtin_amdgcn_s_wait_tensorcnt(0);
    __syncthreads();
  }

  const int lane = t & 31, nl = lane & 15, moff = (lane >> 4) * 8;
#pragma unroll
  for (int mi = 0; mi < 4; ++mi)
#pragma unroll
    for (int ni = 0; ni < 4; ++ni) {
      int col = n0 + wn + 16 * ni + nl;
#pragma unroll
      for (int r = 0; r < 8; ++r) {
        int row = m0 + wm + 16 * mi + moff + r;
        Sb[(size_t)row * 2048 + col] = (_Float16)(acc[mi][ni][r] * 0.03125f); // 1/sqrt(1024)
      }
    }
}

// ---------- kernel 5: in-place row softmax on f16 scores, causal length, zero tail ----------
__global__ __launch_bounds__(256) void ksoftmax(_Float16* __restrict__ Sc) {
  const int q = blockIdx.x, t = threadIdx.x;
  const int len = q + 1;
  _Float16* row = Sc + ((size_t)blockIdx.y * 2048 + q) * 2048;
  __shared__ float red[256];
  float m = -1e30f;
  for (int i = t; i < len; i += 256) m = fmaxf(m, (float)row[i]);
  red[t] = m; __syncthreads();
  for (int s = 128; s > 0; s >>= 1) { if (t < s) red[t] = fmaxf(red[t], red[t + s]); __syncthreads(); }
  m = red[0]; __syncthreads();
  float sum = 0.f;
  for (int i = t; i < len; i += 256) { float e = __expf((float)row[i] - m); sum += e; row[i] = (_Float16)e; }
  red[t] = sum; __syncthreads();
  for (int s = 128; s > 0; s >>= 1) { if (t < s) red[t] += red[t + s]; __syncthreads(); }
  const float inv = 1.0f / red[0];
  for (int i = t; i < len; i += 256) row[i] = (_Float16)((float)row[i] * inv);
  for (int i = len + t; i < 2048; i += 256) row[i] = (_Float16)0.f;   // dense P*V safe
}

// ---------- kernel 6: contexts = P V, both tiles TDM double-buffered ----------
__global__ __launch_bounds__(128) void kgemm_av(const _Float16* __restrict__ P,
                                                const _Float16* __restrict__ Vt,
                                                float* __restrict__ Obase) {
  __shared__ __align__(16) _Float16 lA[2][128 * 32];
  __shared__ __align__(16) _Float16 lB[2][128 * 32];
  const _Float16* Pb  = P  + (size_t)blockIdx.z * 2048 * 2048;  // [q][kv]
  const _Float16* Vtb = Vt + (size_t)blockIdx.z * 1024 * 2048;  // [d][kv]
  float* Ob = Obase + (size_t)blockIdx.z * 2048 * 1024;
  const int n0 = blockIdx.x * 128;   // d
  const int m0 = blockIdx.y * 128;   // q
  const int t = threadIdx.x;
  const int w = t >> 5, wm = (w & 1) * 64, wn = (w >> 1) * 64;
  const bool issuer = (t < 32);
  const int kend = m0 + 128;         // P is zero above diagonal within this range

  if (issuer) {
    tdm_load_tile_f16(lds_off(&lA[0][0]), &Pb[(size_t)m0 * 2048], 2048, 2048, 2048, 32, 128);
    tdm_load_tile_f16(lds_off(&lB[0][0]), &Vtb[(size_t)n0 * 2048], 2048, 1024, 2048, 32, 128);
    __builtin_amdgcn_s_wait_tensorcnt(0);
  }
  __syncthreads();

  v8f acc[4][4] = {};
  for (int kv0 = 0; kv0 < kend; kv0 += 32) {
    const int cur = (kv0 >> 5) & 1, nxt = cur ^ 1;
    if (issuer && (kv0 + 32 < kend)) {
      tdm_load_tile_f16(lds_off(&lA[nxt][0]), &Pb[(size_t)m0 * 2048 + kv0 + 32], 2048, 2048, 2048, 32, 128);
      tdm_load_tile_f16(lds_off(&lB[nxt][0]), &Vtb[(size_t)n0 * 2048 + kv0 + 32], 2048, 1024, 2048, 32, 128);
    }
    mma_step(lA[cur], lB[cur], acc, wm, wn);
    if (issuer && (kv0 + 32 < kend)) __builtin_amdgcn_s_wait_tensorcnt(0);
    __syncthreads();
  }

  const int lane = t & 31, nl = lane & 15, moff = (lane >> 4) * 8;
#pragma unroll
  for (int mi = 0; mi < 4; ++mi)
#pragma unroll
    for (int ni = 0; ni < 4; ++ni) {
      int col = n0 + wn + 16 * ni + nl;
#pragma unroll
      for (int r = 0; r < 8; ++r) {
        int row = m0 + wm + 16 * mi + moff + r;
        Ob[(size_t)row * 1024 + col] = acc[mi][ni][r];
      }
    }
}

// ---------- host-side orchestration ----------
extern "C" void kernel_launch(void* const* d_in, const int* in_sizes, int n_in,
                              void* d_out, int out_size, void* d_ws, size_t ws_size,
                              hipStream_t stream) {
  (void)in_sizes; (void)n_in; (void)out_size; (void)ws_size;
  const float* X  = (const float*)d_in[0];
  const float* Wq = (const float*)d_in[1];
  const float* Wk = (const float*)d_in[2];
  const float* Wv = (const float*)d_in[3];
  float* out = (float*)d_out;
  char* ws = (char*)d_ws;

  const size_t MB = 1024u * 1024u;
  // Phase 1 layout                      // Phase 2 layout (overlays dead phase-1 data)
  _Float16* Xh = (_Float16*)(ws + 0);          // 16 MB, dead after QKV
  _Float16* Wt = (_Float16*)(ws + 16 * MB);    //  6 MB, dead after QKV
  _Float16* Sc = (_Float16*)(ws + 0);          // 32 MB f16 scores->weights [4][2048][2048]
  _Float16* Qh = (_Float16*)(ws + 32 * MB);    // 16 MB [b*2048+q][d]
  _Float16* Kh = (_Float16*)(ws + 48 * MB);    // 16 MB [b*2048+kv][d]
  _Float16* Vt = (_Float16*)(ws + 64 * MB);    // 16 MB [b][d][kv]  (pre-transposed)
  // total scratch: 80 MB

  // 1) X -> f16
  kconvert_x<<<8192, 256, 0, stream>>>(X, Xh, 8192 * 1024 / 4);
  // 2) W_{q,k,v} -> transposed f16
  kconvert_wT<<<dim3(32, 32, 3), 256, 0, stream>>>(Wq, Wk, Wv, Wt);
  // 3) Q,K row-major + V transposed
  kgemm_qkv<<<dim3(8, 64, 3), 128, 0, stream>>>(Xh, Wt, Qh, Vt);
  // 4) scores for all batches (f16, scaled)
  kgemm_scores<<<dim3(16, 16, 4), 128, 0, stream>>>(Qh, Kh, Sc);
  // 5) in-place causal softmax
  ksoftmax<<<dim3(2048, 4), 256, 0, stream>>>(Sc);
  // 6) contexts = P V
  kgemm_av<<<dim3(8, 16, 4), 128, 0, stream>>>(Sc, Vt, out);
}
